// MambaBlock_30648886624685
// MI455X (gfx1250) — matrix operand order repaired and checked by reading, not compiled, not run on hardware
//
#include <hip/hip_runtime.h>
#include <stdint.h>

// ---------------- problem constants ----------------
#define BATCH   2
#define SEQLEN  1024
#define NTOK    (BATCH * SEQLEN)          // 2048
#define DMODEL  768
#define DINNER  1536
#define NHEADS  24
#define HEADDIM 64
#define DSTATE  64
#define DCONV   4
#define CONVDIM (DINNER + 2 * DSTATE)     // 1664
#define DPROJ   (2 * DINNER + 2 * DSTATE + NHEADS)  // 3224
#define NPAD    3264                      // 3224 padded to 51*64
#define DT_OFF  (2 * DINNER + 2 * DSTATE) // 3200
#define EPSV    1e-5f

// ---------------- WMMA fragment types ----------------
typedef __attribute__((ext_vector_type(16))) __bf16   bf16x16;
typedef __attribute__((ext_vector_type(8)))  float    f32x8;
typedef __attribute__((ext_vector_type(8)))  uint16_t u16x8;

union Frag {
  struct { u16x8 lo, hi; } h;   // elements 0..7 / 8..15
  bf16x16 v;
};

struct FragSet {                // one 32(M)x64(N)x32(K) slice of operands
  Frag a0, a1;                  // A rows m0..m0+15 / m0+16..m0+31
  Frag b0, b1, b2, b3;          // B cols n0+0/16/32/48
};

__device__ __forceinline__ uint16_t f2bf(float f) {
  uint32_t u = __float_as_uint(f);
  uint32_t r = u + 0x7FFFu + ((u >> 16) & 1u);   // round-to-nearest-even
  return (uint16_t)(r >> 16);
}

__device__ __forceinline__ f32x8 wmma_bf16(bf16x16 a, bf16x16 b, f32x8 c) {
  // (neg_a, A, neg_b, B, c_mod, C, reuse_a, reuse_b)
  return __builtin_amdgcn_wmma_f32_16x16x32_bf16(false, a, false, b, (short)0, c,
                                                 false, false);
}

__device__ __forceinline__ float sigmoidf_(float x) { return 1.f / (1.f + expf(-x)); }

// ---------------- pipelined 32x64 WMMA GEMM ----------------
// A  [M][K]  bf16 row-major (activations)
// Bt [N][K]  bf16 (weights pre-transposed; fragment loads are contiguous)
// C  [M][ldc] f32 ;  optional residual added on store.
template <int K>
__device__ __forceinline__ void loadA(Frag& f, const uint16_t* __restrict__ base, int kk, int half) {
  f.h.lo = *(const u16x8*)(base + kk + 16 * half);      // K = kk + 16*half + [0,8)
  f.h.hi = *(const u16x8*)(base + kk + 16 * half + 8);  // K = kk + 16*half + 8 + [0,8)
}
__device__ __forceinline__ void loadB(Frag& f, const uint16_t* __restrict__ base, int kk) {
  f.h.lo = *(const u16x8*)(base + kk);                  // K = kk + 16*half + [0,16)
  f.h.hi = *(const u16x8*)(base + kk + 8);
}

template <int K>
__global__ void __launch_bounds__(256)
gemm_bf16_kernel(const uint16_t* __restrict__ A, const uint16_t* __restrict__ Bt,
                 const float* __restrict__ resid, float* __restrict__ C,
                 int nGroups, int ldc) {
  int wave = (blockIdx.x * 256 + threadIdx.x) >> 5;
  int lane = threadIdx.x & 31;
  int mt = wave / nGroups;                 // 32-row group
  int ng = wave - mt * nGroups;            // 64-col group
  int m0 = mt << 5, n0 = ng << 6;
  int half = lane >> 4, l16 = lane & 15;

  const uint16_t* arow0 = A  + (size_t)(m0 + l16) * K;
  const uint16_t* arow1 = arow0 + (size_t)16 * K;
  const uint16_t* brow  = Bt + (size_t)(n0 + l16) * K + 16 * half;

  f32x8 acc[2][4];
  #pragma unroll
  for (int i = 0; i < 2; ++i)
    #pragma unroll
    for (int j = 0; j < 4; ++j) acc[i][j] = f32x8{};

  FragSet s0, s1;
  // prologue: k-slice 0
  loadA<K>(s0.a0, arow0, 0, half);  loadA<K>(s0.a1, arow1, 0, half);
  loadB(s0.b0, brow, 0);            loadB(s0.b1, brow + 16 * K, 0);
  loadB(s0.b2, brow + 32 * K, 0);   loadB(s0.b3, brow + 48 * K, 0);

  for (int kk = 0; kk < K; kk += 64) {
    // stage 1: fetch slice kk+32 while computing slice kk
    __builtin_prefetch(brow + kk + 256, 0, 3);          // global_prefetch_b8 (near)
    loadA<K>(s1.a0, arow0, kk + 32, half);  loadA<K>(s1.a1, arow1, kk + 32, half);
    loadB(s1.b0, brow + kk + 32, 0);        loadB(s1.b1, brow + 16 * K + kk + 32, 0);
    loadB(s1.b2, brow + 32 * K + kk + 32, 0); loadB(s1.b3, brow + 48 * K + kk + 32, 0);

    acc[0][0] = wmma_bf16(s0.a0.v, s0.b0.v, acc[0][0]);
    acc[0][1] = wmma_bf16(s0.a0.v, s0.b1.v, acc[0][1]);
    acc[0][2] = wmma_bf16(s0.a0.v, s0.b2.v, acc[0][2]);
    acc[0][3] = wmma_bf16(s0.a0.v, s0.b3.v, acc[0][3]);
    acc[1][0] = wmma_bf16(s0.a1.v, s0.b0.v, acc[1][0]);
    acc[1][1] = wmma_bf16(s0.a1.v, s0.b1.v, acc[1][1]);
    acc[1][2] = wmma_bf16(s0.a1.v, s0.b2.v, acc[1][2]);
    acc[1][3] = wmma_bf16(s0.a1.v, s0.b3.v, acc[1][3]);

    // stage 2: fetch slice kk+64 while computing slice kk+32
    if (kk + 64 < K) {
      loadA<K>(s0.a0, arow0, kk + 64, half);  loadA<K>(s0.a1, arow1, kk + 64, half);
      loadB(s0.b0, brow + kk + 64, 0);        loadB(s0.b1, brow + 16 * K + kk + 64, 0);
      loadB(s0.b2, brow + 32 * K + kk + 64, 0); loadB(s0.b3, brow + 48 * K + kk + 64, 0);
    }

    acc[0][0] = wmma_bf16(s1.a0.v, s1.b0.v, acc[0][0]);
    acc[0][1] = wmma_bf16(s1.a0.v, s1.b1.v, acc[0][1]);
    acc[0][2] = wmma_bf16(s1.a0.v, s1.b2.v, acc[0][2]);
    acc[0][3] = wmma_bf16(s1.a0.v, s1.b3.v, acc[0][3]);
    acc[1][0] = wmma_bf16(s1.a1.v, s1.b0.v, acc[1][0]);
    acc[1][1] = wmma_bf16(s1.a1.v, s1.b1.v, acc[1][1]);
    acc[1][2] = wmma_bf16(s1.a1.v, s1.b2.v, acc[1][2]);
    acc[1][3] = wmma_bf16(s1.a1.v, s1.b3.v, acc[1][3]);
  }

  int colBase = n0 + l16;
  if (resid) {
    #pragma unroll
    for (int i = 0; i < 2; ++i)
      #pragma unroll
      for (int r = 0; r < 8; ++r) {
        size_t row = (size_t)(m0 + i * 16 + r + 8 * half) * ldc;
        C[row + colBase]      = acc[i][0][r] + resid[row + colBase];
        C[row + colBase + 16] = acc[i][1][r] + resid[row + colBase + 16];
        C[row + colBase + 32] = acc[i][2][r] + resid[row + colBase + 32];
        C[row + colBase + 48] = acc[i][3][r] + resid[row + colBase + 48];
      }
  } else {
    #pragma unroll
    for (int i = 0; i < 2; ++i)
      #pragma unroll
      for (int r = 0; r < 8; ++r) {
        size_t row = (size_t)(m0 + i * 16 + r + 8 * half) * ldc;
        C[row + colBase]      = acc[i][0][r];
        C[row + colBase + 16] = acc[i][1][r];
        C[row + colBase + 32] = acc[i][2][r];
        C[row + colBase + 48] = acc[i][3][r];
      }
  }
}

// ---------------- weight conversion / transpose ----------------
// W_in [768][3224] (f32, row-major) -> WinT [3264][768] (bf16), pad rows zero
__global__ void __launch_bounds__(256)
cvt_win_kernel(const float* __restrict__ W, uint16_t* __restrict__ Wt) {
  int idx = blockIdx.x * 256 + threadIdx.x;       // over NPAD*DMODEL, exact
  int n = idx / DMODEL;
  int k = idx - n * DMODEL;
  float v = (n < DPROJ) ? W[(size_t)k * DPROJ + n] : 0.0f;
  Wt[idx] = f2bf(v);
}

// W_out [1536][768] -> WoutT [768][1536] (bf16)
__global__ void __launch_bounds__(256)
cvt_wout_kernel(const float* __restrict__ W, uint16_t* __restrict__ Wt) {
  int idx = blockIdx.x * 256 + threadIdx.x;       // over DMODEL*DINNER, exact
  int n = idx / DINNER;
  int k = idx - n * DINNER;
  Wt[idx] = f2bf(W[(size_t)k * DMODEL + n]);
}

// ---------------- LayerNorm -> bf16 activations ----------------
__global__ void __launch_bounds__(256)
ln_kernel(const float* __restrict__ x, const float* __restrict__ ln_w,
          const float* __restrict__ ln_b, uint16_t* __restrict__ xn_bf) {
  int token = blockIdx.x;
  const float* row = x + (size_t)token * DMODEL;
  __shared__ float s1[256], s2[256];
  float a = 0.f, b = 0.f;
  for (int i = threadIdx.x; i < DMODEL; i += 256) {
    float v = row[i]; a += v; b += v * v;
  }
  s1[threadIdx.x] = a; s2[threadIdx.x] = b;
  __syncthreads();
  for (int off = 128; off > 0; off >>= 1) {
    if (threadIdx.x < off) {
      s1[threadIdx.x] += s1[threadIdx.x + off];
      s2[threadIdx.x] += s2[threadIdx.x + off];
    }
    __syncthreads();
  }
  float mu   = s1[0] * (1.f / DMODEL);
  float var  = s2[0] * (1.f / DMODEL) - mu * mu;
  float rstd = rsqrtf(var + EPSV);
  for (int i = threadIdx.x; i < DMODEL; i += 256) {
    float v = (row[i] - mu) * rstd * ln_w[i] + ln_b[i];
    xn_bf[(size_t)token * DMODEL + i] = f2bf(v);
  }
}

// ---------------- depthwise conv + SiLU + dt/decay ----------------
__global__ void __launch_bounds__(256)
conv_dt_kernel(const float* __restrict__ zx, const float* __restrict__ conv_w,
               const float* __restrict__ conv_b, const float* __restrict__ dt_bias,
               const float* __restrict__ A_log, float* __restrict__ xconv,
               float* __restrict__ dt, float* __restrict__ decay) {
  int token = blockIdx.x;
  int b = token / SEQLEN, l = token - b * SEQLEN;
  for (int c = threadIdx.x; c < CONVDIM; c += 256) {
    float acc = conv_b[c];
    #pragma unroll
    for (int k = 0; k < DCONV; ++k) {
      int ls = l - (DCONV - 1) + k;
      if (ls >= 0)
        acc += zx[(size_t)(b * SEQLEN + ls) * NPAD + DINNER + c] * conv_w[c * DCONV + k];
    }
    acc = acc * sigmoidf_(acc);                   // SiLU
    xconv[(size_t)token * CONVDIM + c] = acc;
  }
  if (threadIdx.x < NHEADS) {
    int hh = threadIdx.x;
    float raw = zx[(size_t)token * NPAD + DT_OFF + hh] + dt_bias[hh];
    float dtv = (raw > 20.f) ? raw : log1pf(expf(raw));   // softplus
    dt[(size_t)token * NHEADS + hh]    = dtv;
    decay[(size_t)token * NHEADS + hh] = expf(dtv * (-expf(A_log[hh])));
  }
}

// ---------------- sequential SSM scan: one block per (batch, head) ----------------
__global__ void __launch_bounds__(256)
scan_kernel(const float* __restrict__ xconv, const float* __restrict__ dt,
            const float* __restrict__ decay, float* __restrict__ y) {
  int b  = blockIdx.x / NHEADS;
  int hh = blockIdx.x - b * NHEADS;
  int tid = threadIdx.x;
  int p   = tid & 63;          // head-dim element
  int grp = tid >> 6;          // state-dim group (0..3)
  int n0  = grp << 4;
  __shared__ float sB[DSTATE], sC[DSTATE], sX[HEADDIM];
  __shared__ float partial[256];
  float h[16];
  #pragma unroll
  for (int j = 0; j < 16; ++j) h[j] = 0.f;

  for (int l = 0; l < SEQLEN; ++l) {
    int token = b * SEQLEN + l;
    if (tid < 64) {
      sX[tid] = xconv[(size_t)token * CONVDIM + hh * HEADDIM + tid];
      sB[tid] = xconv[(size_t)token * CONVDIM + DINNER + tid];
      sC[tid] = xconv[(size_t)token * CONVDIM + DINNER + DSTATE + tid];
    }
    float dec = decay[(size_t)token * NHEADS + hh];
    float dtv = dt[(size_t)token * NHEADS + hh];
    __syncthreads();
    float coef = dtv * sX[p];
    float acc = 0.f;
    #pragma unroll
    for (int j = 0; j < 16; ++j) {
      h[j] = h[j] * dec + coef * sB[n0 + j];
      acc += h[j] * sC[n0 + j];
    }
    partial[tid] = acc;
    __syncthreads();
    if (tid < 64) {
      y[(size_t)token * DINNER + hh * HEADDIM + tid] =
          partial[tid] + partial[64 + tid] + partial[128 + tid] + partial[192 + tid];
    }
    __syncthreads();
  }
}

// ---------------- gate (y + D*x) * silu(z), RMSNorm -> bf16 ----------------
__global__ void __launch_bounds__(256)
gate_rms_kernel(const float* __restrict__ y, const float* __restrict__ xconv,
                const float* __restrict__ zx, const float* __restrict__ Dp,
                const float* __restrict__ rms_w, uint16_t* __restrict__ g_bf) {
  int token = blockIdx.x;
  const float* yrow = y     + (size_t)token * DINNER;
  const float* xrow = xconv + (size_t)token * CONVDIM;
  const float* zrow = zx    + (size_t)token * NPAD;
  float g[DINNER / 256];
  float ss = 0.f;
  #pragma unroll
  for (int j = 0; j < DINNER / 256; ++j) {
    int d = threadIdx.x + j * 256;
    float yv = yrow[d] + Dp[d >> 6] * xrow[d];
    float z  = zrow[d];
    float gv = yv * (z * sigmoidf_(z));
    g[j] = gv; ss += gv * gv;
  }
  __shared__ float red[256];
  red[threadIdx.x] = ss;
  __syncthreads();
  for (int off = 128; off > 0; off >>= 1) {
    if (threadIdx.x < off) red[threadIdx.x] += red[threadIdx.x + off];
    __syncthreads();
  }
  float rstd = rsqrtf(red[0] * (1.f / DINNER) + EPSV);
  #pragma unroll
  for (int j = 0; j < DINNER / 256; ++j) {
    int d = threadIdx.x + j * 256;
    g_bf[(size_t)token * DINNER + d] = f2bf(g[j] * rstd * rms_w[d]);
  }
}

// ---------------- host launcher ----------------
extern "C" void kernel_launch(void* const* d_in, const int* in_sizes, int n_in,
                              void* d_out, int out_size, void* d_ws, size_t ws_size,
                              hipStream_t stream) {
  const float* x      = (const float*)d_in[0];
  const float* ln_w   = (const float*)d_in[1];
  const float* ln_b   = (const float*)d_in[2];
  const float* W_in   = (const float*)d_in[3];
  const float* conv_w = (const float*)d_in[4];
  const float* conv_b = (const float*)d_in[5];
  const float* dt_b   = (const float*)d_in[6];
  const float* A_log  = (const float*)d_in[7];
  const float* Dp     = (const float*)d_in[8];
  const float* rms_w  = (const float*)d_in[9];
  const float* W_out  = (const float*)d_in[10];
  float* out = (float*)d_out;

  char* ws = (char*)d_ws;
  size_t off = 0;
  auto alloc = [&](size_t bytes) -> void* {
    void* p = ws + off;
    off = (off + bytes + 255) & ~(size_t)255;
    return p;
  };
  uint16_t* xn_bf  = (uint16_t*)alloc((size_t)NTOK * DMODEL * 2);
  uint16_t* winT   = (uint16_t*)alloc((size_t)NPAD * DMODEL * 2);
  uint16_t* woutT  = (uint16_t*)alloc((size_t)DMODEL * DINNER * 2);
  float*    zx     = (float*)   alloc((size_t)NTOK * NPAD * 4);
  float*    xconv  = (float*)   alloc((size_t)NTOK * CONVDIM * 4);
  float*    dt     = (float*)   alloc((size_t)NTOK * NHEADS * 4);
  float*    decay  = (float*)   alloc((size_t)NTOK * NHEADS * 4);
  float*    y      = (float*)   alloc((size_t)NTOK * DINNER * 4);
  uint16_t* g_bf   = (uint16_t*)alloc((size_t)NTOK * DINNER * 2);
  (void)ws_size; (void)in_sizes; (void)n_in; (void)out_size;

  // 1. weight prep (bf16 + transpose)
  cvt_win_kernel <<<(NPAD * DMODEL) / 256, 256, 0, stream>>>(W_in, winT);
  cvt_wout_kernel<<<(DMODEL * DINNER) / 256, 256, 0, stream>>>(W_out, woutT);
  // 2. layernorm -> bf16 activations
  ln_kernel<<<NTOK, 256, 0, stream>>>(x, ln_w, ln_b, xn_bf);
  // 3. input projection GEMM (WMMA bf16, 32x64 tiles, pipelined)
  //    waves = (2048/32) * (3264/64) = 64*51 = 3264 -> 408 blocks of 8 waves
  gemm_bf16_kernel<DMODEL><<<(64 * (NPAD / 64)) / 8, 256, 0, stream>>>(
      xn_bf, winT, nullptr, zx, NPAD / 64, NPAD);
  // 4. depthwise conv + silu + dt/decay
  conv_dt_kernel<<<NTOK, 256, 0, stream>>>(zx, conv_w, conv_b, dt_b, A_log,
                                           xconv, dt, decay);
  // 5. sequential SSM scan, one block per (batch, head)
  scan_kernel<<<BATCH * NHEADS, 256, 0, stream>>>(xconv, dt, decay, y);
  // 6. gate + RMSNorm -> bf16
  gate_rms_kernel<<<NTOK, 256, 0, stream>>>(y, xconv, zx, Dp, rms_w, g_bf);
  // 7. output projection GEMM (WMMA bf16) + residual
  //    waves = 64 * (768/64) = 768 -> 96 blocks
  gemm_bf16_kernel<DINNER><<<(64 * (DMODEL / 64)) / 8, 256, 0, stream>>>(
      g_bf, woutT, x, out, DMODEL / 64, DMODEL);
}
